// S6Layer_2723009265878
// MI455X (gfx1250) — compile-verified, matrix operand order checked by reading, not run on hardware
//
#include <hip/hip_runtime.h>
#include <hip/hip_bf16.h>
#include <math.h>

// ---------------- problem constants ----------------
#define BATCH   2
#define SEQ     2048
#define D_MODEL 1024
#define D_INNER 2048
#define D_STATE 16
#define DT_RANK 64
#define D_CONV  4
#define NROW    (BATCH * SEQ)          // 4096 token rows
#define XZ_COLS (2 * D_INNER)          // 4096

typedef __attribute__((ext_vector_type(16))) __bf16 v16bf;
typedef __attribute__((ext_vector_type(8)))  __bf16 v8bf;
typedef __attribute__((ext_vector_type(8)))  float  v8f;

// ---------------- f32 -> bf16 cast ----------------
__global__ __launch_bounds__(256)
void cast_f32_bf16_kernel(const float* __restrict__ src, __bf16* __restrict__ dst, int n) {
    int i = blockIdx.x * blockDim.x + threadIdx.x;
    if (i < n) dst[i] = (__bf16)src[i];
}

// ---------------- generic bf16 WMMA GEMM: C[M,N] = A[M,K] * B[N,K]^T ----------------
// block = 256 threads (8 waves) arranged 2(M) x 4(N); block tile 128(M) x 256(N).
// Wave tile: 64(M) x 64(N) = 4x4 grid of 16x16 wmma tiles -> 16 WMMAs per 16 b128 loads.
// A-fragment (16x32 bf16): lane l<16 -> row l, K {0..7,16..23}; lane>=16 -> row l-16, K {8..15,24..31}.
// B-fragment (32x16 bf16): lane -> col (l&15), same K chunking -> contiguous loads from (N,K) row-major.
__global__ __launch_bounds__(256)
void wmma_gemm_bf16_kernel(const __bf16* __restrict__ A,
                           const __bf16* __restrict__ B,
                           float* __restrict__ C,
                           int M, int N, int K)
{
    const int lane = threadIdx.x & 31;
    const int wave = threadIdx.x >> 5;
    const int wm   = wave >> 2;           // 2 waves along M
    const int wn   = wave & 3;            // 4 waves along N
    const int rowBase = blockIdx.y * 128 + wm * 64;
    const int colBase = blockIdx.x * 256 + wn * 64;

    const int l15   = lane & 15;
    const int lhalf = lane >> 4;          // 0 or 1
    const int kSub  = lhalf * 8;          // K sub-offset inside a 32-chunk

    v8f acc[4][4];
#pragma unroll
    for (int i = 0; i < 4; ++i)
#pragma unroll
        for (int j = 0; j < 4; ++j) {
            v8f z = {};
            acc[i][j] = z;
        }

    const __bf16* aBase = A + (size_t)(rowBase + l15) * K + kSub;
    const __bf16* bBase = B + (size_t)(colBase + l15) * K + kSub;

    for (int kk = 0; kk < K; kk += 32) {
        if (kk + 32 < K) {   // prefetch next K-chunk (global_prefetch_b8)
            __builtin_prefetch(aBase + kk + 32, 0, 1);
            __builtin_prefetch(bBase + kk + 32, 0, 1);
        }
        v16bf af[4], bfr[4];
#pragma unroll
        for (int i = 0; i < 4; ++i) {
            const __bf16* p = aBase + (size_t)(i * 16) * K + kk;
            v8bf c0 = *(const v8bf*)(p);
            v8bf c1 = *(const v8bf*)(p + 16);
#pragma unroll
            for (int e = 0; e < 8; ++e) { af[i][e] = c0[e]; af[i][e + 8] = c1[e]; }
        }
#pragma unroll
        for (int j = 0; j < 4; ++j) {
            const __bf16* p = bBase + (size_t)(j * 16) * K + kk;
            v8bf c0 = *(const v8bf*)(p);
            v8bf c1 = *(const v8bf*)(p + 16);
#pragma unroll
            for (int e = 0; e < 8; ++e) { bfr[j][e] = c0[e]; bfr[j][e + 8] = c1[e]; }
        }
#pragma unroll
        for (int i = 0; i < 4; ++i)
#pragma unroll
            for (int j = 0; j < 4; ++j)
                acc[i][j] = __builtin_amdgcn_wmma_f32_16x16x32_bf16(
                    false, af[i], false, bfr[j], (short)0, acc[i][j], false, false);
    }

    // C/D layout: VGPR r -> row r + 8*lhalf, col l15
#pragma unroll
    for (int i = 0; i < 4; ++i)
#pragma unroll
        for (int j = 0; j < 4; ++j) {
            const int col = colBase + j * 16 + l15;
#pragma unroll
            for (int r = 0; r < 8; ++r) {
                const int row = rowBase + i * 16 + lhalf * 8 + r;
                C[(size_t)row * N + col] = acc[i][j][r];
            }
        }
}

// ---------------- causal depthwise conv (k=4) + SiLU ----------------
__global__ __launch_bounds__(256)
void conv_silu_kernel(const float* __restrict__ xz, const float* __restrict__ cw,
                      const float* __restrict__ cb, float* __restrict__ xc)
{
    int idx = blockIdx.x * blockDim.x + threadIdx.x;       // over NROW * D_INNER
    if (idx >= NROW * D_INNER) return;
    int d   = idx & (D_INNER - 1);
    int row = idx >> 11;                                   // b*SEQ + t
    int t   = row & (SEQ - 1);
    int b   = row >> 11;
    float s = cb[d];
#pragma unroll
    for (int k = 0; k < D_CONV; ++k) {
        int tt = t - (D_CONV - 1) + k;
        if (tt >= 0)
            s += xz[(size_t)(b * SEQ + tt) * XZ_COLS + d] * cw[d * D_CONV + k];
    }
    xc[idx] = s / (1.0f + expf(-s));                       // silu
}

// ---------------- x_dbl[row, 0..95] = xc[row,:] @ x_proj_w^T ----------------
// 256 threads: two 128-thread groups split the K=2048 dot in half (float4 LDS reads),
// then partials are combined through LDS.
__global__ __launch_bounds__(256)
void xdbl_kernel(const float* __restrict__ xc, const float* __restrict__ xpw,
                 float* __restrict__ xdbl)
{
    __shared__ float srow[D_INNER];
    __shared__ float psum[2][96];
    const int row = blockIdx.x;
    const float4* xr4 = (const float4*)(xc + (size_t)row * D_INNER);
    for (int i = threadIdx.x; i < D_INNER / 4; i += 256)
        ((float4*)srow)[i] = xr4[i];
    __syncthreads();
    const int g = threadIdx.x >> 7;        // K-half: 0 or 1
    const int j = threadIdx.x & 127;       // output index within group
    if (j < 96) {
        const float4* wr = (const float4*)(xpw + (size_t)j * D_INNER + g * (D_INNER / 2));
        const float4* sr = (const float4*)(srow + g * (D_INNER / 2));
        float s = 0.f;
#pragma unroll 4
        for (int q = 0; q < D_INNER / 8; ++q) {
            float4 a = sr[q], w = wr[q];
            s += a.x * w.x + a.y * w.y + a.z * w.z + a.w * w.w;
        }
        psum[g][j] = s;
    }
    __syncthreads();
    if (threadIdx.x < 96)
        xdbl[(size_t)row * 96 + threadIdx.x] = psum[0][threadIdx.x] + psum[1][threadIdx.x];
}

// ---------------- dt = softplus(dt_r @ dt_proj_w^T + b) ----------------
__global__ __launch_bounds__(256)
void dt_kernel(const float* __restrict__ xdbl, const float* __restrict__ dtw,
               const float* __restrict__ dtb, float* __restrict__ dt)
{
    __shared__ float r64[DT_RANK];
    const int row = blockIdx.y;
    const int d   = blockIdx.x * 256 + threadIdx.x;
    if (threadIdx.x < DT_RANK) r64[threadIdx.x] = xdbl[(size_t)row * 96 + threadIdx.x];
    __syncthreads();
    const float4* w4 = (const float4*)(dtw + (size_t)d * DT_RANK);
    const float4* r4 = (const float4*)r64;
    float s = dtb[d];
#pragma unroll
    for (int r = 0; r < DT_RANK / 4; ++r) {
        float4 a = r4[r], w = w4[r];
        s += a.x * w.x + a.y * w.y + a.z * w.z + a.w * w.w;
    }
    float sp = (s > 20.f) ? s : log1pf(expf(s));
    dt[(size_t)row * D_INNER + d] = sp;
}

// ---------------- selective scan ----------------
// lane layout per wave: lanes 0..15 -> channel c0, state n=lane; lanes 16..31 -> channel c1, n=lane-16.
// 16-lane shuffle reduction over the state dimension produces y each step.
__global__ __launch_bounds__(256)
void scan_kernel(const float* __restrict__ dt, const float* __restrict__ xc,
                 const float* __restrict__ xdbl, const float* __restrict__ A_log,
                 float* __restrict__ y)
{
    const int lane = threadIdx.x & 31;
    const int wave = threadIdx.x >> 5;
    const int n    = lane & 15;
    const int half = lane >> 4;
    const int chan = blockIdx.x * 16 + wave * 2 + half;    // 0 .. BATCH*D_INNER-1
    const int b    = chan / D_INNER;
    const int d    = chan % D_INNER;

    const float Adn = -expf(A_log[d * D_STATE + n]);
    float h = 0.f;
    const size_t rowBase = (size_t)b * SEQ;

    for (int t = 0; t < SEQ; ++t) {
        const size_t row = rowBase + t;
        float dtv = 0.f, xv = 0.f;
        if (n == 0) {
            dtv = dt[row * D_INNER + d];
            xv  = xc[row * D_INNER + d];
        }
        dtv = __shfl(dtv, half * 16, 32);
        xv  = __shfl(xv,  half * 16, 32);
        const float Bv = xdbl[row * 96 + DT_RANK + n];
        const float Cv = xdbl[row * 96 + DT_RANK + D_STATE + n];
        h = expf(dtv * Adn) * h + (dtv * xv) * Bv;
        float p = h * Cv;
        p += __shfl_xor(p, 1, 32);
        p += __shfl_xor(p, 2, 32);
        p += __shfl_xor(p, 4, 32);
        p += __shfl_xor(p, 8, 32);
        if (n == 0) y[row * D_INNER + d] = p;
    }
}

// ---------------- gating: ybf = bf16((y + D*xc) * silu(z)) ----------------
__global__ __launch_bounds__(256)
void gate_kernel(const float* __restrict__ y, const float* __restrict__ xc,
                 const float* __restrict__ xz, const float* __restrict__ Dv,
                 __bf16* __restrict__ ybf)
{
    int idx = blockIdx.x * blockDim.x + threadIdx.x;
    if (idx >= NROW * D_INNER) return;
    int d = idx & (D_INNER - 1);
    size_t row = (size_t)(idx >> 11);
    float z = xz[row * XZ_COLS + D_INNER + d];
    float val = (y[idx] + Dv[d] * xc[idx]) * (z / (1.0f + expf(-z)));
    ybf[idx] = (__bf16)val;
}

// ---------------- host-side orchestration ----------------
extern "C" void kernel_launch(void* const* d_in, const int* in_sizes, int n_in,
                              void* d_out, int out_size, void* d_ws, size_t ws_size,
                              hipStream_t stream)
{
    (void)in_sizes; (void)n_in; (void)out_size; (void)ws_size;
    const float* x          = (const float*)d_in[0];
    const float* in_proj_w  = (const float*)d_in[1];
    const float* conv_w     = (const float*)d_in[2];
    const float* conv_b     = (const float*)d_in[3];
    const float* x_proj_w   = (const float*)d_in[4];
    const float* dt_proj_w  = (const float*)d_in[5];
    const float* dt_proj_b  = (const float*)d_in[6];
    const float* A_log      = (const float*)d_in[7];
    const float* Dvec       = (const float*)d_in[8];
    const float* out_proj_w = (const float*)d_in[9];

    char* ws = (char*)d_ws;
    size_t off = 0;
    auto alloc = [&](size_t bytes) -> void* {
        void* p = ws + off;
        off += (bytes + 255) & ~(size_t)255;
        return p;
    };

    __bf16* x_bf     = (__bf16*)alloc((size_t)NROW * D_MODEL * 2);        // 8.4 MB
    __bf16* w_in_bf  = (__bf16*)alloc((size_t)XZ_COLS * D_MODEL * 2);     // 8.4 MB
    __bf16* w_out_bf = (__bf16*)alloc((size_t)D_MODEL * D_INNER * 2);     // 4.2 MB
    float*  xz       = (float*) alloc((size_t)NROW * XZ_COLS * 4);        // 67 MB
    float*  xc       = (float*) alloc((size_t)NROW * D_INNER * 4);        // 33.5 MB
    float*  xdbl     = (float*) alloc((size_t)NROW * 96 * 4);             // 1.6 MB
    float*  dt       = (float*) alloc((size_t)NROW * D_INNER * 4);        // 33.5 MB
    float*  yb       = (float*) alloc((size_t)NROW * D_INNER * 4);        // 33.5 MB
    __bf16* y_bf     = (__bf16*)alloc((size_t)NROW * D_INNER * 2);        // 16.8 MB

    // 1) casts to bf16
    {
        int n1 = NROW * D_MODEL;
        cast_f32_bf16_kernel<<<(n1 + 255) / 256, 256, 0, stream>>>(x, x_bf, n1);
        int n2 = XZ_COLS * D_MODEL;
        cast_f32_bf16_kernel<<<(n2 + 255) / 256, 256, 0, stream>>>(in_proj_w, w_in_bf, n2);
        int n3 = D_MODEL * D_INNER;
        cast_f32_bf16_kernel<<<(n3 + 255) / 256, 256, 0, stream>>>(out_proj_w, w_out_bf, n3);
    }

    // 2) in_proj GEMM: xz[4096,4096] = x[4096,1024] @ in_proj_w^T
    wmma_gemm_bf16_kernel<<<dim3(XZ_COLS / 256, NROW / 128), 256, 0, stream>>>(
        x_bf, w_in_bf, xz, NROW, XZ_COLS, D_MODEL);

    // 3) causal conv + silu
    conv_silu_kernel<<<(NROW * D_INNER + 255) / 256, 256, 0, stream>>>(xz, conv_w, conv_b, xc);

    // 4) x_dbl = xc @ x_proj_w^T   (96 outputs per row)
    xdbl_kernel<<<NROW, 256, 0, stream>>>(xc, x_proj_w, xdbl);

    // 5) dt = softplus(dt_r @ dt_proj_w^T + dt_proj_b)
    dt_kernel<<<dim3(D_INNER / 256, NROW), 256, 0, stream>>>(xdbl, dt_proj_w, dt_proj_b, dt);

    // 6) selective scan
    scan_kernel<<<(BATCH * D_INNER) / 16, 256, 0, stream>>>(dt, xc, xdbl, A_log, yb);

    // 7) gating + cast
    gate_kernel<<<(NROW * D_INNER + 255) / 256, 256, 0, stream>>>(yb, xc, xz, Dvec, y_bf);

    // 8) out_proj GEMM: out[4096,1024] = y[4096,2048] @ out_proj_w^T
    wmma_gemm_bf16_kernel<<<dim3(D_MODEL / 256, NROW / 128), 256, 0, stream>>>(
        y_bf, w_out_bf, (float*)d_out, NROW, D_MODEL, D_INNER);
}
